// SyntaxGCN_12506944766171
// MI455X (gfx1250) — matrix operand order repaired and checked by reading, not compiled, not run on hardware
//
#include <hip/hip_runtime.h>
#include <hip/hip_bf16.h>
#include <math.h>

#define IN_DIM 512
#define HID 256

typedef __attribute__((ext_vector_type(16))) __bf16 v16bf;
typedef __attribute__((ext_vector_type(8)))  float  v8f;

// ---------- bf16 bit helpers (round-to-nearest-even) ----------
__device__ __forceinline__ unsigned short f2bfu(float f) {
    union { float f; unsigned u; } v; v.f = f;
    unsigned r = v.u + 0x7FFFu + ((v.u >> 16) & 1u);
    return (unsigned short)(r >> 16);
}
__device__ __forceinline__ float bfu2f(unsigned short s) {
    union { unsigned u; float f; } v; v.u = ((unsigned)s) << 16;
    return v.f;
}

// ---------- degree init: self-loop contributes 1 to every node ----------
__global__ void k_deg_init(float* deg, int N) {
    int i = blockIdx.x * blockDim.x + threadIdx.x;
    if (i < N) deg[i] = 1.0f;
}

// ---------- degree accumulation over edge destinations ----------
__global__ void k_deg_accum(const int* __restrict__ ei, float* deg, int E) {
    int e = blockIdx.x * blockDim.x + threadIdx.x;
    if (e < E) atomicAdd(&deg[ei[E + e]], 1.0f);
}

// ---------- dinv = rsqrt(deg) in place ----------
__global__ void k_dinv(float* deg, int N) {
    int i = blockIdx.x * blockDim.x + threadIdx.x;
    if (i < N) deg[i] = rsqrtf(deg[i]);
}

// ---------- pack x into fragment-ordered bf16 hi/lo (A fragments, 16x32 layout) ----------
// dst index t = ((tm*16 + kc)*32 + lane)*16 + j ; j = 2v+p
// kk = ((v>>2)<<4) + ((lane>>4)<<3) + ((v&3)<<1) + p ; row = tm*16 + (lane&15)
__global__ void k_pack_x(const float* __restrict__ x, unsigned short* __restrict__ hi,
                         unsigned short* __restrict__ lo, int N, int total) {
    int t = blockIdx.x * blockDim.x + threadIdx.x;
    if (t >= total) return;
    const int j    = t & 15;
    const int lane = (t >> 4) & 31;
    const int kc   = (t >> 9) & 15;
    const int tm   = t >> 13;
    const int v = j >> 1, p = j & 1;
    const int kk = ((v >> 2) << 4) + ((lane >> 4) << 3) + ((v & 3) << 1) + p;
    const int row = tm * 16 + (lane & 15);
    const float f = (row < N) ? x[(size_t)row * IN_DIM + kc * 32 + kk] : 0.0f;
    const unsigned short uh = f2bfu(f);
    hi[t] = uh;
    lo[t] = f2bfu(f - bfu2f(uh));
}

// ---------- pack W into fragment-ordered bf16 hi/lo (B fragments, 32x16 layout) ----------
// dst index t = ((tn*16 + kc)*32 + lane)*16 + j ; j = 2v+p
// kk = ((v>>2)<<4) + ((lane>>4)<<3) + ((v&3)<<1) + p ; col = tn*16 + (lane&15)
__global__ void k_pack_w(const float* __restrict__ W, unsigned short* __restrict__ hi,
                         unsigned short* __restrict__ lo, int total) {
    int t = blockIdx.x * blockDim.x + threadIdx.x;
    if (t >= total) return;
    const int j    = t & 15;
    const int lane = (t >> 4) & 31;
    const int kc   = (t >> 9) & 15;
    const int tn   = t >> 13;
    const int v = j >> 1, p = j & 1;
    const int kk  = ((v >> 2) << 4) + ((lane >> 4) << 3) + ((v & 3) << 1) + p;
    const int col = tn * 16 + (lane & 15);
    const float f = W[(size_t)(kc * 32 + kk) * HID + col];
    const unsigned short uh = f2bfu(f);
    hi[t] = uh;
    lo[t] = f2bfu(f - bfu2f(uh));
}

// ---------- h = x @ W via WMMA bf16x3 split, pre-packed fragments ----------
// One wave computes a 16x64 strip (4 col-tiles); A fragments reused 4x.
__global__ void __launch_bounds__(256) k_gemm_wmma(const unsigned short* __restrict__ Ah,
                                                  const unsigned short* __restrict__ Al,
                                                  const unsigned short* __restrict__ Bh,
                                                  const unsigned short* __restrict__ Bl,
                                                  float* __restrict__ h, int tiles_m) {
    const int lane  = threadIdx.x & 31;
    const int wave  = threadIdx.x >> 5;
    const int gwave = blockIdx.x * 8 + wave;
    if (gwave >= tiles_m * 4) return;

    const int tm  = gwave >> 2;       // row tile
    const int tn0 = (gwave & 3) * 4;  // first of 4 col tiles

    const v16bf* __restrict__ pAh = (const v16bf*)Ah;
    const v16bf* __restrict__ pAl = (const v16bf*)Al;
    const v16bf* __restrict__ pBh = (const v16bf*)Bh;
    const v16bf* __restrict__ pBl = (const v16bf*)Bl;

    v8f acc[4] = {};

#pragma unroll 4
    for (int kc = 0; kc < 16; ++kc) {
        const v16bf a_hi = pAh[(tm * 16 + kc) * 32 + lane];
        const v16bf a_lo = pAl[(tm * 16 + kc) * 32 + lane];
#pragma unroll
        for (int t = 0; t < 4; ++t) {
            const int bi = ((tn0 + t) * 16 + kc) * 32 + lane;
            const v16bf b_hi = pBh[bi];
            const v16bf b_lo = pBl[bi];
            acc[t] = __builtin_amdgcn_wmma_f32_16x16x32_bf16(false, a_hi, false, b_hi,
                                                             (short)0, acc[t], false, false);
            acc[t] = __builtin_amdgcn_wmma_f32_16x16x32_bf16(false, a_lo, false, b_hi,
                                                             (short)0, acc[t], false, false);
            acc[t] = __builtin_amdgcn_wmma_f32_16x16x32_bf16(false, a_hi, false, b_lo,
                                                             (short)0, acc[t], false, false);
        }
    }

    // store: C/D 16x16 f32 layout -> M = r + 8*(lane>>4), N = lane&15
    const int half = lane >> 4;
    const int l15  = lane & 15;
    const int row0 = tm * 16;
#pragma unroll
    for (int t = 0; t < 4; ++t) {
        const int col = (tn0 + t) * 16 + l15;
#pragma unroll
        for (int r = 0; r < 8; ++r) {
            h[(size_t)(row0 + r + (half << 3)) * HID + col] = acc[t][r];
        }
    }
}

// ---------- edge scatter: agg[dst] += h[src] * dinv[src]*dinv[dst]  (wave per edge) ----------
__global__ void __launch_bounds__(256) k_edge_scatter(const int* __restrict__ ei,
                                                      const float* __restrict__ h,
                                                      const float* __restrict__ dinv,
                                                      float* __restrict__ agg, int E) {
    const int wid  = (int)((blockIdx.x * blockDim.x + threadIdx.x) >> 5);
    const int lane = threadIdx.x & 31;
    if (wid >= E) return;
    const int s = ei[wid];
    const int d = ei[E + wid];
    const float norm = dinv[s] * dinv[d];
    const float4* __restrict__ hs = (const float4*)(h + (size_t)s * HID);
    float* __restrict__ ad = agg + (size_t)d * HID;
#pragma unroll
    for (int j = 0; j < 2; ++j) {
        float4 v = hs[lane * 2 + j];
        const int c = lane * 8 + j * 4;
        atomicAdd(ad + c + 0, v.x * norm);
        atomicAdd(ad + c + 1, v.y * norm);
        atomicAdd(ad + c + 2, v.z * norm);
        atomicAdd(ad + c + 3, v.w * norm);
    }
}

// ---------- self-loop + bias + relu, then pool into per-graph sums ----------
__global__ void k_node_finish(const float* __restrict__ agg, const float* __restrict__ h,
                              const float* __restrict__ dinv, const float* __restrict__ b,
                              const int* __restrict__ batch, float* __restrict__ sums,
                              int N) {
    const int i = blockIdx.x;
    const int c = threadIdx.x;
    if (i >= N) return;
    const float di = dinv[i];
    float val = agg[(size_t)i * HID + c] + h[(size_t)i * HID + c] * di * di + b[c];
    val = fmaxf(val, 0.0f);
    atomicAdd(&sums[(size_t)batch[i] * HID + c], val);
}

// ---------- per-graph node counts ----------
__global__ void k_counts(const int* __restrict__ batch, float* counts, int N) {
    int i = blockIdx.x * blockDim.x + threadIdx.x;
    if (i < N) atomicAdd(&counts[batch[i]], 1.0f);
}

// ---------- head: out[g] = sigmoid(mean(xr_g) . lin_w + lin_b) ----------
__global__ void k_head(const float* __restrict__ sums, const float* __restrict__ counts,
                       const float* __restrict__ lin_w, const float* __restrict__ lin_b,
                       float* __restrict__ out, int G) {
    __shared__ float red[HID];
    const int g = blockIdx.x;
    const int c = threadIdx.x;
    if (g >= G) return;
    const float cnt = fmaxf(counts[g], 1.0f);
    red[c] = (sums[(size_t)g * HID + c] / cnt) * lin_w[c];
    __syncthreads();
    for (int s = HID / 2; s > 0; s >>= 1) {
        if (c < s) red[c] += red[c + s];
        __syncthreads();
    }
    if (c == 0) out[g] = 1.0f / (1.0f + expf(-(red[0] + lin_b[0])));
}

extern "C" void kernel_launch(void* const* d_in, const int* in_sizes, int n_in,
                              void* d_out, int out_size, void* d_ws, size_t ws_size,
                              hipStream_t stream) {
    const float* x     = (const float*)d_in[0];   // [N, 512]
    const int*   ei    = (const int*)d_in[1];     // [2, E] flat
    const int*   batch = (const int*)d_in[2];     // [N]
    const float* W     = (const float*)d_in[3];   // [512, 256]
    const float* b     = (const float*)d_in[4];   // [256]
    const float* lin_w = (const float*)d_in[5];   // [256]
    const float* lin_b = (const float*)d_in[6];   // [1]
    float*       out   = (float*)d_out;           // [G]

    const int N = in_sizes[0] / IN_DIM;
    const int E = in_sizes[1] / 2;
    const int G = out_size;
    const int tiles_m = (N + 15) / 16;

    // packed fragment sizes (in elements)
    const int packA = tiles_m * (IN_DIM / 32) * 32 * 16;  // per-lane 16 bf16 per chunk
    const int packB = (HID / 16) * (IN_DIM / 32) * 32 * 16;

    // workspace carve-out
    float* h      = (float*)d_ws;                      // N*HID f32
    float* agg    = h + (size_t)N * HID;               // N*HID f32
    float* dinv   = agg + (size_t)N * HID;             // N f32 (deg then dinv)
    float* sums   = dinv + N;                          // G*HID f32
    float* counts = sums + (size_t)G * HID;            // G f32
    unsigned short* xh = (unsigned short*)(counts + G);
    unsigned short* xl = xh + (size_t)packA;
    unsigned short* wh = xl + (size_t)packA;
    unsigned short* wl = wh + (size_t)packB;

    // zero accumulators
    hipMemsetAsync(agg, 0, (size_t)N * HID * sizeof(float), stream);
    hipMemsetAsync(sums, 0, (size_t)G * HID * sizeof(float), stream);
    hipMemsetAsync(counts, 0, (size_t)G * sizeof(float), stream);

    // degrees -> dinv
    k_deg_init<<<(N + 255) / 256, 256, 0, stream>>>(dinv, N);
    k_deg_accum<<<(E + 255) / 256, 256, 0, stream>>>(ei, dinv, E);
    k_dinv<<<(N + 255) / 256, 256, 0, stream>>>(dinv, N);

    // pack fragments (one-time hi/lo bf16 split, fragment-ordered)
    k_pack_x<<<(packA + 255) / 256, 256, 0, stream>>>(x, xh, xl, N, packA);
    k_pack_w<<<(packB + 255) / 256, 256, 0, stream>>>(W, wh, wl, packB);

    // GEMM h = x @ W  (wave per 16x64 strip, 8 waves per block)
    const int nwaves = tiles_m * 4;
    k_gemm_wmma<<<(nwaves + 7) / 8, 256, 0, stream>>>(xh, xl, wh, wl, h, tiles_m);

    // edge scatter (wave per edge, 8 waves per block)
    k_edge_scatter<<<(E + 7) / 8, 256, 0, stream>>>(ei, h, dinv, agg, E);

    // self-loop + bias + relu + pool
    k_node_finish<<<N, HID, 0, stream>>>(agg, h, dinv, b, batch, sums, N);
    k_counts<<<(N + 255) / 256, 256, 0, stream>>>(batch, counts, N);

    // head
    k_head<<<G, HID, 0, stream>>>(sums, counts, lin_w, lin_b, out, G);
}
